// RL2LSTM_1030792151351
// MI455X (gfx1250) — matrix operand order repaired and checked by reading, not compiled, not run on hardware
//
#include <hip/hip_runtime.h>
#include <hip/hip_bf16.h>
#include <math.h>

typedef __bf16 bf16;
typedef __attribute__((ext_vector_type(16))) __bf16 v16bf;
typedef __attribute__((ext_vector_type(8)))  __bf16 v8bf;
typedef __attribute__((ext_vector_type(8)))  float  v8f;

#define T_STEPS 128
#define BATCH   512
#define HID     256
#define GATES   1024
#define ENC_RAW 834
#define ENC_PAD 864   // 27 * 32, K padded for 16x16x32 WMMA

union AFrag { v16bf v; v8bf h[2]; };

static __device__ __forceinline__ v8f wmma_bf16(v16bf a, v16bf b, v8f c) {
  // D = A(16x32 bf16) * B(32x16 bf16) + C(16x16 f32)
  return __builtin_amdgcn_wmma_f32_16x16x32_bf16(false, a, false, b, (short)0, c,
                                                 false, false);
}

// ---- Pack a row-major f32 [Ksrc, N] weight into WMMA B-fragment order -------
// Block b = ntile*(Kpad/32) + ktile holds 512 bf16: lane*16 + e, where
// h = lane/16: e<8 -> K = ktile*32 + 8h + e,   e>=8 -> K = ktile*32 + 16 + 8h + (e-8),
// column n = ntile*16 + (lane&15). K >= Ksrc pads with zero.
__global__ void k_pack(const float* __restrict__ src, bf16* __restrict__ dst,
                       int Ksrc, int Kpad, int N) {
  int idx = blockIdx.x * blockDim.x + threadIdx.x;
  int total = Kpad * N;
  if (idx >= total) return;
  int nKt    = Kpad >> 5;
  int block  = idx >> 9;
  int within = idx & 511;
  int lane   = within >> 4;
  int e      = within & 15;
  int ktile  = block % nKt;
  int ntile  = block / nKt;
  int K = ktile * 32 + ((e >= 8) ? 8 : 0) + ((lane >= 16) ? 8 : 0) + e;
  int n = ntile * 16 + (lane & 15);
  float v = (K < Ksrc) ? src[K * N + n] : 0.0f;
  dst[idx] = (bf16)v;
}

// ---- State init: c, h_f32, hseq[slot 0] -------------------------------------
__global__ void k_init(const float* __restrict__ c0, const float* __restrict__ h0,
                       float* __restrict__ cbuf, float* __restrict__ h32,
                       bf16* __restrict__ hseq) {
  int i = blockIdx.x * blockDim.x + threadIdx.x;
  if (i >= BATCH * HID) return;
  cbuf[i] = c0[i];
  h32[i]  = h0[i];
  hseq[i] = (bf16)h0[i];
}

// ---- Fused: encode -> x = relu(enc@W1+b1) -> xg = x@Wx ----------------------
// One block = 16 rows (t*B+b), 128 threads = 4 waves.
__global__ __launch_bounds__(128) void k_front(
    const int* __restrict__ obs_img, const float* __restrict__ obs_dir,
    const int* __restrict__ pact, const float* __restrict__ prew,
    const float* __restrict__ done,
    const float* __restrict__ tile_emb, const float* __restrict__ color_emb,
    const float* __restrict__ act_emb, const float* __restrict__ Wdir,
    const float* __restrict__ bdir, const float* __restrict__ b1,
    const bf16* __restrict__ w1p, const bf16* __restrict__ wxp,
    bf16* __restrict__ xg)
{
  __shared__ bf16 enc[16 * ENC_PAD];   // 27.0 KB
  __shared__ bf16 xs [16 * HID];       //  8.0 KB
  const int tid  = threadIdx.x;
  const int row0 = blockIdx.x * 16;

  // Phase 1: gather 16 encoded rows into LDS (bf16, zero-padded to 864)
  {
    int i = tid >> 3;          // row within tile
    int j = tid & 7;
    int row = row0 + i;        // global (t*B + b)
    for (int c = j; c < ENC_PAD; c += 8) {
      float v;
      if (c < 800) {
        int p = c >> 5;        // 5x5 grid position
        int w = c & 31;
        if (w < 16) {
          int tix = obs_img[(row * 25 + p) * 2 + 0];
          v = tile_emb[tix * 16 + w];
        } else {
          int cix = obs_img[(row * 25 + p) * 2 + 1];
          v = color_emb[cix * 16 + (w - 16)];
        }
      } else if (c < 816) {          // dir_enc = obs_dir @ Wdir + bdir
        int d = c - 800;
        float s = bdir[d];
        #pragma unroll
        for (int k = 0; k < 4; ++k) s += obs_dir[row * 4 + k] * Wdir[k * 16 + d];
        v = s;
      } else if (c < 832) {
        v = act_emb[pact[row] * 16 + (c - 816)];
      } else if (c == 832) {
        v = prew[row];
      } else if (c == 833) {
        v = done[row];
      } else {
        v = 0.0f;                    // K padding
      }
      enc[i * ENC_PAD + c] = (bf16)v;
    }
  }
  __syncthreads();

  const int wv        = tid >> 5;
  const int lane      = tid & 31;
  const int rowInTile = lane & 15;
  const int hiHalf    = lane >> 4;   // A-frag K-chunk selector

  // Phase 2: x = relu(enc @ W1 + b1); each wave owns 4 column tiles
  {
    v8f acc[4] = {};
    for (int k = 0; k < ENC_PAD / 32; ++k) {
      AFrag a;
      const bf16* ap = &enc[rowInTile * ENC_PAD + k * 32 + hiHalf * 8];
      a.h[0] = *(const v8bf*)(ap);
      a.h[1] = *(const v8bf*)(ap + 16);
      #pragma unroll
      for (int q = 0; q < 4; ++q) {
        int nt = wv * 4 + q;
        v16bf b = *(const v16bf*)(w1p + (size_t)(nt * 27 + k) * 512 + lane * 16);
        acc[q] = wmma_bf16(a.v, b, acc[q]);
      }
    }
    #pragma unroll
    for (int q = 0; q < 4; ++q) {
      int col = (wv * 4 + q) * 16 + (lane & 15);
      float bb = b1[col];
      #pragma unroll
      for (int r = 0; r < 8; ++r) {
        int i = r + hiHalf * 8;     // C-layout: vgpr r -> M=r (+8 for hi lanes)
        xs[i * HID + col] = (bf16)fmaxf(acc[q][r] + bb, 0.0f);
      }
    }
  }
  __syncthreads();

  // Phase 3: xg = x @ Wx; each wave owns 16 of the 64 gate-column tiles
  for (int q = 0; q < 16; ++q) {
    int nt = wv * 16 + q;
    v8f acc = {};
    for (int k = 0; k < HID / 32; ++k) {
      AFrag a;
      const bf16* ap = &xs[rowInTile * HID + k * 32 + hiHalf * 8];
      a.h[0] = *(const v8bf*)(ap);
      a.h[1] = *(const v8bf*)(ap + 16);
      v16bf b = *(const v16bf*)(wxp + (size_t)(nt * 8 + k) * 512 + lane * 16);
      acc = wmma_bf16(a.v, b, acc);
    }
    int col = nt * 16 + (lane & 15);
    #pragma unroll
    for (int r = 0; r < 8; ++r) {
      int i = r + hiHalf * 8;
      xg[(size_t)(row0 + i) * GATES + col] = (bf16)acc[r];
    }
  }
}

// ---- One LSTM timestep: z = xg_t + h@Wh + bh; gates; update c,h -------------
// 512 waves total: wave -> (mtile 0..31, nt 0..15); 4 gate tiles per wave.
__global__ __launch_bounds__(128) void k_lstm_step(
    int t, const bf16* __restrict__ xg, const bf16* __restrict__ whp,
    const float* __restrict__ bh, float* __restrict__ cbuf,
    float* __restrict__ h32, bf16* __restrict__ hseq)
{
  const int tid   = threadIdx.x;
  const int wv    = tid >> 5;
  const int lane  = tid & 31;
  const int wgl   = blockIdx.x * 4 + wv;   // 0..511
  const int mtile = wgl >> 4;              // 0..31
  const int nt    = wgl & 15;              // hidden-column tile
  const int rowInTile = lane & 15;
  const int hiHalf    = lane >> 4;
  const bf16* hprev = hseq + (size_t)t * (BATCH * HID);

  v8f acc[4] = {};
  for (int k = 0; k < HID / 32; ++k) {
    AFrag a;
    const bf16* ap = hprev + (mtile * 16 + rowInTile) * HID + k * 32 + hiHalf * 8;
    a.h[0] = *(const v8bf*)ap;
    a.h[1] = *(const v8bf*)(ap + 16);
    #pragma unroll
    for (int g = 0; g < 4; ++g) {
      int ntile = g * 16 + nt;             // gate g occupies cols g*256..
      const bf16* bp = whp + (size_t)(ntile * 8 + k) * 512 + lane * 16;
      __builtin_prefetch(bp + 512, 0, 1);  // next k's fragment -> global_prefetch_b8
      v16bf b = *(const v16bf*)bp;
      acc[g] = wmma_bf16(a.v, b, acc[g]);
    }
  }

  const int col = nt * 16 + (lane & 15);
  const bf16* xgt = xg + (size_t)t * (BATCH * GATES);
  #pragma unroll
  for (int r = 0; r < 8; ++r) {
    int row = mtile * 16 + r + hiHalf * 8;
    float zi = acc[0][r] + (float)xgt[(size_t)row * GATES + 0 * HID + col] + bh[0 * HID + col];
    float zf = acc[1][r] + (float)xgt[(size_t)row * GATES + 1 * HID + col] + bh[1 * HID + col];
    float zg = acc[2][r] + (float)xgt[(size_t)row * GATES + 2 * HID + col] + bh[2 * HID + col];
    float zo = acc[3][r] + (float)xgt[(size_t)row * GATES + 3 * HID + col] + bh[3 * HID + col];
    float si = 1.0f / (1.0f + __expf(-zi));
    float sf = 1.0f / (1.0f + __expf(-zf));
    float so = 1.0f / (1.0f + __expf(-zo));
    float tg = tanhf(zg);
    int ci = row * HID + col;
    float cn = sf * cbuf[ci] + si * tg;
    cbuf[ci] = cn;
    float hn = so * tanhf(cn);
    h32[ci] = hn;
    hseq[(size_t)(t + 1) * (BATCH * HID) + ci] = (bf16)hn;
  }
}

// ---- Policy / value heads (memory-bound; VALU dot) --------------------------
__global__ void k_head(const bf16* __restrict__ hseq, const float* __restrict__ Wpi,
                       const float* __restrict__ bpi, const float* __restrict__ Wv,
                       const float* __restrict__ bv, float* __restrict__ out)
{
  int idx = blockIdx.x * blockDim.x + threadIdx.x;
  if (idx >= T_STEPS * BATCH * 7) return;
  int row = idx / 7;
  int o   = idx - row * 7;
  const bf16* h = hseq + (size_t)(row + BATCH) * HID;  // slot t+1 = h after step t
  float s = 0.0f;
  if (o < 6) {
    for (int k = 0; k < HID; ++k) s += (float)h[k] * Wpi[k * 6 + o];
    out[(size_t)row * 6 + o] = s + bpi[o];
  } else {
    for (int k = 0; k < HID; ++k) s += (float)h[k] * Wv[k];
    out[(size_t)(T_STEPS * BATCH * 6) + row] = s + bv[0];
  }
}

// ---- Final cf / hf copies ---------------------------------------------------
__global__ void k_tail(const float* __restrict__ cbuf, const float* __restrict__ h32,
                       float* __restrict__ out) {
  int i = blockIdx.x * blockDim.x + threadIdx.x;
  if (i >= BATCH * HID) return;
  out[(size_t)T_STEPS * BATCH * 7 + i] = cbuf[i];
  out[(size_t)T_STEPS * BATCH * 7 + BATCH * HID + i] = h32[i];
}

extern "C" void kernel_launch(void* const* d_in, const int* in_sizes, int n_in,
                              void* d_out, int out_size, void* d_ws, size_t ws_size,
                              hipStream_t stream) {
  (void)in_sizes; (void)n_in; (void)out_size; (void)ws_size;
  const int*   obs_img   = (const int*)  d_in[0];
  const float* obs_dir   = (const float*)d_in[1];
  const int*   pact      = (const int*)  d_in[2];
  const float* prew      = (const float*)d_in[3];
  const float* done      = (const float*)d_in[4];
  const float* c0        = (const float*)d_in[5];
  const float* h0        = (const float*)d_in[6];
  const float* tile_emb  = (const float*)d_in[7];
  const float* color_emb = (const float*)d_in[8];
  const float* act_emb   = (const float*)d_in[9];
  const float* Wdir      = (const float*)d_in[10];
  const float* bdir      = (const float*)d_in[11];
  const float* W1        = (const float*)d_in[12];
  const float* b1        = (const float*)d_in[13];
  const float* Wx        = (const float*)d_in[14];
  const float* Wh        = (const float*)d_in[15];
  const float* bh        = (const float*)d_in[16];
  const float* Wpi       = (const float*)d_in[17];
  const float* bpi       = (const float*)d_in[18];
  const float* Wv        = (const float*)d_in[19];
  const float* bv        = (const float*)d_in[20];
  float* out = (float*)d_out;

  // Workspace layout (bytes, all 256-aligned):
  char* ws = (char*)d_ws;
  bf16*  w1p  = (bf16*) (ws + 0);           // 864*256*2   = 442368
  bf16*  wxp  = (bf16*) (ws + 442368);      // 256*1024*2  = 524288
  bf16*  whp  = (bf16*) (ws + 966656);      // 256*1024*2  = 524288
  bf16*  xg   = (bf16*) (ws + 1490944);     // 65536*1024*2 = 134217728
  bf16*  hseq = (bf16*) (ws + 135708672);   // 129*512*256*2 = 33816576
  float* cbuf = (float*)(ws + 169525248);   // 512*256*4   = 524288
  float* h32  = (float*)(ws + 170049536);   // 512*256*4   = 524288

  k_pack<<<(864 * 256 + 255) / 256, 256, 0, stream>>>(W1, w1p, ENC_RAW, ENC_PAD, HID);
  k_pack<<<(256 * 1024 + 255) / 256, 256, 0, stream>>>(Wx, wxp, HID, HID, GATES);
  k_pack<<<(256 * 1024 + 255) / 256, 256, 0, stream>>>(Wh, whp, HID, HID, GATES);
  k_init<<<(BATCH * HID + 255) / 256, 256, 0, stream>>>(c0, h0, cbuf, h32, hseq);

  k_front<<<(T_STEPS * BATCH) / 16, 128, 0, stream>>>(
      obs_img, obs_dir, pact, prew, done, tile_emb, color_emb, act_emb,
      Wdir, bdir, b1, w1p, wxp, xg);

  for (int t = 0; t < T_STEPS; ++t)
    k_lstm_step<<<128, 128, 0, stream>>>(t, xg, whp, bh, cbuf, h32, hseq);

  k_head<<<(T_STEPS * BATCH * 7 + 255) / 256, 256, 0, stream>>>(
      hseq, Wpi, bpi, Wv, bv, out);
  k_tail<<<(BATCH * HID + 255) / 256, 256, 0, stream>>>(cbuf, h32, out);
}
